// GroupedQueryAttention_47949014892588
// MI455X (gfx1250) — matrix operand order repaired and checked by reading, not compiled, not run on hardware
//
#include <hip/hip_runtime.h>

typedef _Float16 half_t;
typedef __attribute__((ext_vector_type(16))) _Float16 v16h;
typedef __attribute__((ext_vector_type(8)))  float    v8f;
typedef __attribute__((ext_vector_type(4)))  int      v4i;

union Frag16 {
    v16h h;
    uint4 q[2];
};

// ---------------------------------------------------------------------------
// CDNA5 async global->LDS copy (ASYNCcnt-tracked), with sync fallback.
// ---------------------------------------------------------------------------
#if defined(__has_builtin)
#if __has_builtin(__builtin_amdgcn_global_load_async_to_lds_b128)
#define HAS_ASYNC_LDS 1
#endif
#endif
#ifndef HAS_ASYNC_LDS
#define HAS_ASYNC_LDS 0
#endif

__device__ __forceinline__ void copy_b128_to_lds(const half_t* gsrc, half_t* ldst) {
#if HAS_ASYNC_LDS
    __builtin_amdgcn_global_load_async_to_lds_b128(
        (__attribute__((address_space(1))) v4i*)(uintptr_t)gsrc,
        (__attribute__((address_space(3))) v4i*)(unsigned)(uintptr_t)ldst,
        0, 0);
#else
    *(uint4*)ldst = *(const uint4*)gsrc;
#endif
}

__device__ __forceinline__ void async_copies_wait() {
#if HAS_ASYNC_LDS
#if __has_builtin(__builtin_amdgcn_s_wait_asynccnt)
    __builtin_amdgcn_s_wait_asynccnt(0);
#else
    asm volatile("s_wait_asynccnt 0x0" ::: "memory");
#endif
#endif
}

// ---------------------------------------------------------------------------
// fp32 -> f16 conversion, 4 elements / thread
// ---------------------------------------------------------------------------
__global__ __launch_bounds__(256) void cvt_f32_to_f16_x4(const float4* __restrict__ in,
                                                         uint2* __restrict__ out, int n4) {
    int i = blockIdx.x * blockDim.x + threadIdx.x;
    if (i < n4) {
        float4 f = in[i];
        half_t h[4] = {(half_t)f.x, (half_t)f.y, (half_t)f.z, (half_t)f.w};
        out[i] = *(uint2*)h;
    }
}

// ---------------------------------------------------------------------------
// Double-buffered WMMA GEMM:  C[M,N] = A[M,K] f16 * B[K,N] f16 + bias
// MODE 0: f32 row-major.   MODE 1: f16 scatter to [B, HH, S=2048, 64].
// Block tile 128x128, BK=32, 256 threads = 8 waves; wave tile 64x32.
// One barrier per K-step; async A-tile DMA overlaps WMMA compute.
// ---------------------------------------------------------------------------
#define BM 128
#define BN 128
#define BKK 32
#define LDA 40   // LDS row stride in halfs

template <int MODE, int HH>
__global__ __launch_bounds__(256) void gemm_wmma_f16(
        const half_t* __restrict__ A, const half_t* __restrict__ Bw,
        const float* __restrict__ bias,
        float* __restrict__ outF, half_t* __restrict__ outH,
        int M, int N, int K) {
    constexpr int SSHIFT = 11;        // S = 2048
    constexpr int SMASK  = 2047;
    __shared__ half_t Alds[2][BM * LDA];
    __shared__ half_t Btl[2][BN * LDA];   // B transposed: [n][k]

    const int tid  = threadIdx.x;
    const int lane = tid & 31;
    const int wid  = tid >> 5;
    const int sel  = lane >> 4;
    const int ln   = lane & 15;

    const int bm = blockIdx.y * BM;
    const int bn = blockIdx.x * BN;
    const int wm = (wid >> 2) * 64;
    const int wn = (wid & 3) * 32;

    v8f acc[4][2];
#pragma unroll
    for (int mt = 0; mt < 4; ++mt)
#pragma unroll
        for (int nt = 0; nt < 2; ++nt)
            acc[mt][nt] = (v8f){0.f, 0.f, 0.f, 0.f, 0.f, 0.f, 0.f, 0.f};

    // staging thread roles
    const int arow = tid >> 1, ac0 = (tid & 1) * 16;        // A tile
    const int bkr  = (tid >> 4) * 2, bn0 = (tid & 15) * 8;  // B tile (k-row pair)

    const half_t* aptr  = A + (size_t)(bm + arow) * K + ac0;
    const half_t* bptr0 = Bw + (size_t)bkr * N + bn + bn0;

    // ---- prologue: async A(0), register-prefetch B(0) ----
    copy_b128_to_lds(aptr,     &Alds[0][arow * LDA + ac0]);
    copy_b128_to_lds(aptr + 8, &Alds[0][arow * LDA + ac0 + 8]);
    uint4 br0 = *(const uint4*)bptr0;
    uint4 br1 = *(const uint4*)(bptr0 + N);

    const int nsteps = K / BKK;
    for (int i = 0; i < nsteps; ++i) {
        const int cur = i & 1, nxt = cur ^ 1;

        // (a) commit B registers, transposed + pair-packed, into LDS[cur]
        {
            half_t a0[8], a1[8];
            *(uint4*)a0 = br0;
            *(uint4*)a1 = br1;
#pragma unroll
            for (int t = 0; t < 8; ++t) {
                union { half_t h[2]; unsigned u; } pk;
                pk.h[0] = a0[t];
                pk.h[1] = a1[t];
                *(unsigned*)&Btl[cur][(bn0 + t) * LDA + bkr] = pk.u;
            }
        }
        // (b) current A tile DMA done, (c) staging visible to all waves
        async_copies_wait();
        __syncthreads();

        // (d) prefetch tile i+1 (overlaps compute below)
        if (i + 1 < nsteps) {
            const half_t* an = aptr + (size_t)(i + 1) * BKK;
            copy_b128_to_lds(an,     &Alds[nxt][arow * LDA + ac0]);
            copy_b128_to_lds(an + 8, &Alds[nxt][arow * LDA + ac0 + 8]);
            const half_t* bp = bptr0 + (size_t)(i + 1) * BKK * N;
            br0 = *(const uint4*)bp;
            br1 = *(const uint4*)(bp + N);
        }

        // (e) compute from buffers [cur]
        Frag16 bfr[2];
#pragma unroll
        for (int nt = 0; nt < 2; ++nt) {
            const half_t* bp = &Btl[cur][(wn + nt * 16 + ln) * LDA];
            bfr[nt].q[0] = *(const uint4*)(bp + sel * 8);
            bfr[nt].q[1] = *(const uint4*)(bp + 16 + sel * 8);
        }
#pragma unroll
        for (int mt = 0; mt < 4; ++mt) {
            Frag16 afr;
            const half_t* ap = &Alds[cur][(wm + mt * 16 + ln) * LDA];
            afr.q[0] = *(const uint4*)(ap + sel * 8);
            afr.q[1] = *(const uint4*)(ap + 16 + sel * 8);
#pragma unroll
            for (int nt = 0; nt < 2; ++nt)
                acc[mt][nt] = __builtin_amdgcn_wmma_f32_16x16x32_f16(
                    false, afr.h, false, bfr[nt].h, (short)0, acc[mt][nt], false, false);
        }
    }

    // --- epilogue ---
#pragma unroll
    for (int mt = 0; mt < 4; ++mt) {
#pragma unroll
        for (int nt = 0; nt < 2; ++nt) {
            int col = bn + wn + nt * 16 + ln;
            float bv = bias[col];
#pragma unroll
            for (int j = 0; j < 8; ++j) {
                int row = bm + wm + mt * 16 + j + sel * 8;
                float v = acc[mt][nt][j] + bv;
                if (MODE == 0) {
                    outF[(size_t)row * N + col] = v;
                } else {
                    int b = row >> SSHIFT, s = row & SMASK;
                    int h = col >> 6, d = col & 63;
                    outH[(((size_t)(b * HH + h)) << SSHIFT | s) * 64 + d] = (half_t)v;
                }
            }
        }
    }
}

// ---------------------------------------------------------------------------
// Double-buffered flash-style GQA attention.
// grid.x = B*HQ (64), grid.y = S/128. 256 threads = 8 waves; wave owns 16 q rows.
// ---------------------------------------------------------------------------
#define LKV 72

__global__ __launch_bounds__(256) void gqa_attn_wmma(
        const half_t* __restrict__ Qh, const half_t* __restrict__ Kh,
        const half_t* __restrict__ Vh, half_t* __restrict__ ctx) {
    constexpr int Ss = 2048;
    __shared__ half_t Klds[2][64 * LKV];    // [key][d]
    __shared__ half_t Vt[2][64 * LKV];      // [d][key]
    __shared__ half_t Plds[8 * 16 * LKV];   // per-wave P buffer [row][key]

    const int tid  = threadIdx.x;
    const int lane = tid & 31;
    const int wid  = tid >> 5;
    const int sel  = lane >> 4;
    const int ln   = lane & 15;

    const int bh  = blockIdx.x;
    const int b   = bh >> 5;
    const int h   = bh & 31;
    const int hkv = h >> 2;
    const int q0  = blockIdx.y * 128 + wid * 16;

    const half_t* Qp = Qh + ((size_t)bh * Ss + q0) * 64;
    const half_t* Kp = Kh + ((size_t)(b * 8 + hkv) * Ss) * 64;
    const half_t* Vp = Vh + ((size_t)(b * 8 + hkv) * Ss) * 64;

    Frag16 qf[2];
    {
        const half_t* qrow = Qp + (size_t)ln * 64;
#pragma unroll
        for (int f = 0; f < 2; ++f) {
            qf[f].q[0] = *(const uint4*)(qrow + f * 32 + sel * 8);
            qf[f].q[1] = *(const uint4*)(qrow + f * 32 + 16 + sel * 8);
        }
    }

    v8f acc[4];
#pragma unroll
    for (int nt = 0; nt < 4; ++nt)
        acc[nt] = (v8f){0.f, 0.f, 0.f, 0.f, 0.f, 0.f, 0.f, 0.f};
    float mi[8], li[8];
#pragma unroll
    for (int j = 0; j < 8; ++j) { mi[j] = -1e30f; li[j] = 0.f; }

    half_t* Pw = &Plds[wid * 16 * LKV];
    const float scale = 0.125f;

    // staging roles
    const int kkey = tid >> 2, kd0 = (tid & 3) * 16;       // K tile (async)
    const int vk2 = (tid & 31) * 2, vd0 = (tid >> 5) * 8;  // V transpose (key pair)

    const half_t* kptr = Kp + (size_t)kkey * 64 + kd0;
    const half_t* vptr = Vp + (size_t)vk2 * 64 + vd0;

    // ---- prologue: async K(0), register-prefetch V(0) ----
    copy_b128_to_lds(kptr,     &Klds[0][kkey * LKV + kd0]);
    copy_b128_to_lds(kptr + 8, &Klds[0][kkey * LKV + kd0 + 8]);
    uint4 vr0 = *(const uint4*)vptr;
    uint4 vr1 = *(const uint4*)(vptr + 64);

    const int nblk = Ss / 64;
    for (int i = 0; i < nblk; ++i) {
        const int cur = i & 1, nxt = cur ^ 1;

        // (a) commit V registers, transposed + pair-packed, into Vt[cur]
        {
            half_t a0[8], a1[8];
            *(uint4*)a0 = vr0;
            *(uint4*)a1 = vr1;
#pragma unroll
            for (int t = 0; t < 8; ++t) {
                union { half_t h[2]; unsigned u; } pk;
                pk.h[0] = a0[t];
                pk.h[1] = a1[t];
                *(unsigned*)&Vt[cur][(vd0 + t) * LKV + vk2] = pk.u;
            }
        }
        // (b) + (c)
        async_copies_wait();
        __syncthreads();

        // (d) prefetch block i+1 (overlaps compute below)
        if (i + 1 < nblk) {
            const half_t* kn = kptr + (size_t)(i + 1) * 64 * 64;
            copy_b128_to_lds(kn,     &Klds[nxt][kkey * LKV + kd0]);
            copy_b128_to_lds(kn + 8, &Klds[nxt][kkey * LKV + kd0 + 8]);
            const half_t* vn = vptr + (size_t)(i + 1) * 64 * 64;
            vr0 = *(const uint4*)vn;
            vr1 = *(const uint4*)(vn + 64);
        }

        // (e) compute: scores S = Q * K^T * scale
        float s[4][8];
#pragma unroll
        for (int nt = 0; nt < 4; ++nt) {
            const half_t* kp = &Klds[cur][(nt * 16 + ln) * LKV];
            v8f sc = (v8f){0.f, 0.f, 0.f, 0.f, 0.f, 0.f, 0.f, 0.f};
#pragma unroll
            for (int f = 0; f < 2; ++f) {
                Frag16 kf;
                kf.q[0] = *(const uint4*)(kp + f * 32 + sel * 8);
                kf.q[1] = *(const uint4*)(kp + f * 32 + 16 + sel * 8);
                sc = __builtin_amdgcn_wmma_f32_16x16x32_f16(
                    false, qf[f].h, false, kf.h, (short)0, sc, false, false);
            }
#pragma unroll
            for (int j = 0; j < 8; ++j) s[nt][j] = sc[j] * scale;
        }

        // online softmax row stats
        float fac[8];
#pragma unroll
        for (int j = 0; j < 8; ++j) {
            float v = fmaxf(fmaxf(s[0][j], s[1][j]), fmaxf(s[2][j], s[3][j]));
            v = fmaxf(v, __shfl_xor(v, 1, 32));
            v = fmaxf(v, __shfl_xor(v, 2, 32));
            v = fmaxf(v, __shfl_xor(v, 4, 32));
            v = fmaxf(v, __shfl_xor(v, 8, 32));
            float mnew = fmaxf(mi[j], v);
            fac[j] = __expf(mi[j] - mnew);
            mi[j] = mnew;
            li[j] *= fac[j];
        }
#pragma unroll
        for (int nt = 0; nt < 4; ++nt)
#pragma unroll
            for (int j = 0; j < 8; ++j) acc[nt][j] *= fac[j];

        // P = exp(S - m) into per-wave LDS; row sums
#pragma unroll
        for (int j = 0; j < 8; ++j) {
            float rsum = 0.f;
            int row = j + sel * 8;
#pragma unroll
            for (int nt = 0; nt < 4; ++nt) {
                float p = __expf(s[nt][j] - mi[j]);
                rsum += p;
                Pw[row * LKV + nt * 16 + ln] = (half_t)p;
            }
            rsum += __shfl_xor(rsum, 1, 32);
            rsum += __shfl_xor(rsum, 2, 32);
            rsum += __shfl_xor(rsum, 4, 32);
            rsum += __shfl_xor(rsum, 8, 32);
            li[j] += rsum;
        }

        // P as A-fragments
        Frag16 pf[2];
        {
            const half_t* prow = &Pw[ln * LKV];
#pragma unroll
            for (int f = 0; f < 2; ++f) {
                pf[f].q[0] = *(const uint4*)(prow + f * 32 + sel * 8);
                pf[f].q[1] = *(const uint4*)(prow + f * 32 + 16 + sel * 8);
            }
        }

        // O += P * V
#pragma unroll
        for (int nt = 0; nt < 4; ++nt) {
            const half_t* vp = &Vt[cur][(nt * 16 + ln) * LKV];
#pragma unroll
            for (int f = 0; f < 2; ++f) {
                Frag16 vf;
                vf.q[0] = *(const uint4*)(vp + f * 32 + sel * 8);
                vf.q[1] = *(const uint4*)(vp + f * 32 + 16 + sel * 8);
                acc[nt] = __builtin_amdgcn_wmma_f32_16x16x32_f16(
                    false, pf[f].h, false, vf.h, (short)0, acc[nt], false, false);
            }
        }
    }

    // --- finalize ---
#pragma unroll
    for (int j = 0; j < 8; ++j) {
        float inv = 1.0f / li[j];
        int qrow = q0 + j + sel * 8;
        half_t* dst = ctx + ((size_t)(b * Ss + qrow)) * 2048 + h * 64;
#pragma unroll
        for (int nt = 0; nt < 4; ++nt)
            dst[nt * 16 + ln] = (half_t)(acc[nt][j] * inv);
    }
}

// ---------------------------------------------------------------------------
// Host launch
// ---------------------------------------------------------------------------
extern "C" void kernel_launch(void* const* d_in, const int* in_sizes, int n_in,
                              void* d_out, int out_size, void* d_ws, size_t ws_size,
                              hipStream_t stream) {
    (void)in_sizes; (void)n_in; (void)out_size; (void)ws_size;

    const float* x   = (const float*)d_in[0];
    const float* W_q = (const float*)d_in[1];
    const float* b_q = (const float*)d_in[2];
    const float* W_k = (const float*)d_in[3];
    const float* b_k = (const float*)d_in[4];
    const float* W_v = (const float*)d_in[5];
    const float* b_v = (const float*)d_in[6];
    const float* W_o = (const float*)d_in[7];
    const float* b_o = (const float*)d_in[8];
    float* out = (float*)d_out;

    const int Bv = 2, Sv = 2048, Dm = 2048, HQv = 32, HKVv = 8;
    const int Mv = Bv * Sv;
    const int NKV = HKVv * 64;

    char* ws = (char*)d_ws;
    size_t off = 0;
    auto walloc = [&](size_t bytes) -> void* {
        void* p = ws + off;
        off += (bytes + 255) & ~(size_t)255;
        return p;
    };
    half_t* xh   = (half_t*)walloc((size_t)Mv * Dm * 2);
    half_t* Wqh  = (half_t*)walloc((size_t)Dm * Dm * 2);
    half_t* Wkh  = (half_t*)walloc((size_t)Dm * NKV * 2);
    half_t* Wvh  = (half_t*)walloc((size_t)Dm * NKV * 2);
    half_t* Woh  = (half_t*)walloc((size_t)Dm * Dm * 2);
    half_t* Qhb  = (half_t*)walloc((size_t)Bv * HQv * Sv * 64 * 2);
    half_t* Khb  = (half_t*)walloc((size_t)Bv * HKVv * Sv * 64 * 2);
    half_t* Vhb  = (half_t*)walloc((size_t)Bv * HKVv * Sv * 64 * 2);
    half_t* ctxh = (half_t*)walloc((size_t)Mv * Dm * 2);

    // 1) conversions
    int n4;
    n4 = Mv * Dm / 4;
    cvt_f32_to_f16_x4<<<n4 / 256, 256, 0, stream>>>((const float4*)x, (uint2*)xh, n4);
    n4 = Dm * Dm / 4;
    cvt_f32_to_f16_x4<<<n4 / 256, 256, 0, stream>>>((const float4*)W_q, (uint2*)Wqh, n4);
    n4 = Dm * NKV / 4;
    cvt_f32_to_f16_x4<<<n4 / 256, 256, 0, stream>>>((const float4*)W_k, (uint2*)Wkh, n4);
    n4 = Dm * NKV / 4;
    cvt_f32_to_f16_x4<<<n4 / 256, 256, 0, stream>>>((const float4*)W_v, (uint2*)Wvh, n4);
    n4 = Dm * Dm / 4;
    cvt_f32_to_f16_x4<<<n4 / 256, 256, 0, stream>>>((const float4*)W_o, (uint2*)Woh, n4);

    // 2) projections
    gemm_wmma_f16<1, 32><<<dim3(Dm / BN, Mv / BM), 256, 0, stream>>>(
        xh, Wqh, b_q, nullptr, Qhb, Mv, Dm, Dm);
    gemm_wmma_f16<1, 8><<<dim3(NKV / BN, Mv / BM), 256, 0, stream>>>(
        xh, Wkh, b_k, nullptr, Khb, Mv, NKV, Dm);
    gemm_wmma_f16<1, 8><<<dim3(NKV / BN, Mv / BM), 256, 0, stream>>>(
        xh, Wvh, b_v, nullptr, Vhb, Mv, NKV, Dm);

    // 3) attention
    gqa_attn_wmma<<<dim3(Bv * HQv, Sv / 128), 256, 0, stream>>>(Qhb, Khb, Vhb, ctxh);

    // 4) output projection
    gemm_wmma_f16<0, 0><<<dim3(Dm / BN, Mv / BM), 256, 0, stream>>>(
        ctxh, Woh, b_o, out, nullptr, Mv, Dm, Dm);
}
